// TransformerModel_20985210208658
// MI455X (gfx1250) — compile-verified, matrix operand order checked by reading, not run on hardware
//
#include <hip/hip_runtime.h>
#include <hip/hip_bf16.h>

// ---------------------------------------------------------------------------
// Model constants (from the reference)
// ---------------------------------------------------------------------------
#define N_LAYER 8
#define N_HEAD 8
#define D_MODEL 512
#define D_K 64
#define D_HIDDEN 2048
#define N_DICT 8192
#define BB 8
#define TT 1024
#define NTOK (BB * TT)   // 8192 tokens

typedef __attribute__((ext_vector_type(16))) __bf16 v16bf;
typedef __attribute__((ext_vector_type(8)))  __bf16 v8bf;
typedef __attribute__((ext_vector_type(8)))  float  v8f;
typedef __attribute__((ext_vector_type(4)))  int    v4i;

// gfx1250 async global->LDS path (guarded; falls back to ds stores)
#if defined(__has_builtin)
#  if __has_builtin(__builtin_amdgcn_global_load_async_to_lds_b128)
#    define HAVE_ASYNC_LDS 1
#  endif
#endif
#ifndef HAVE_ASYNC_LDS
#  define HAVE_ASYNC_LDS 0
#endif

static __device__ __forceinline__ void wait_async0() {
#if HAVE_ASYNC_LDS
#  if __has_builtin(__builtin_amdgcn_s_wait_asynccnt)
  __builtin_amdgcn_s_wait_asynccnt(0);
#  else
  asm volatile("s_wait_asynccnt 0x0" ::: "memory");
#  endif
#endif
}

// float -> bf16 round-to-nearest-even via integer trick
static __device__ __forceinline__ __bf16 f2bf(float f) {
  union { float f; unsigned u; } v; v.f = f;
  unsigned r = (v.u + 0x7FFFu + ((v.u >> 16) & 1u)) >> 16;
  unsigned short s = (unsigned short)r;
  __bf16 out;
  __builtin_memcpy(&out, &s, 2);
  return out;
}

// block-wide sum, blockDim.x == 256
static __device__ __forceinline__ float block_sum256(float v, float* red) {
  int tid = threadIdx.x;
  red[tid] = v;
  __syncthreads();
  for (int s = 128; s > 0; s >>= 1) {
    if (tid < s) red[tid] += red[tid + s];
    __syncthreads();
  }
  float r = red[0];
  __syncthreads();
  return r;
}

// ---------------------------------------------------------------------------
// Weight conversion: fp32 (R x C, row-major, batched) -> bf16 transposed (C x R)
// ---------------------------------------------------------------------------
__global__ void k_transpose_to_bf16(const float* __restrict__ src,
                                    __bf16* __restrict__ dst, int R, int C) {
  int b = blockIdx.z;
  const float* s = src + (size_t)b * R * C;
  __bf16* d = dst + (size_t)b * R * C;
  int c = blockIdx.x * 16 + threadIdx.x;
  int r = blockIdx.y * 16 + threadIdx.y;
  if (r < R && c < C) d[(size_t)c * R + r] = f2bf(s[(size_t)r * C + c]);
}

// ---------------------------------------------------------------------------
// Embedding + positional + LayerNorm.  grid = NTOK blocks, 256 threads.
// ---------------------------------------------------------------------------
__global__ void k_embed_ln(const int* __restrict__ x,
                           const float* __restrict__ emb,
                           const float* __restrict__ pos,
                           float* __restrict__ hf, __bf16* __restrict__ hb) {
  __shared__ float red[256];
  int tok = blockIdx.x;
  int t = tok & (TT - 1);
  int id = x[tok];
  int d0 = threadIdx.x;
  int d1 = threadIdx.x + 256;
  float v0 = emb[(size_t)id * D_MODEL + d0] + pos[(size_t)t * D_MODEL + d0];
  float v1 = emb[(size_t)id * D_MODEL + d1] + pos[(size_t)t * D_MODEL + d1];
  float mu = block_sum256(v0 + v1, red) * (1.0f / D_MODEL);
  float a0 = v0 - mu, a1 = v1 - mu;
  float var = block_sum256(a0 * a0 + a1 * a1, red) * (1.0f / D_MODEL);
  float rs = rsqrtf(var + 1e-5f);
  float o0 = a0 * rs, o1 = a1 * rs;
  size_t base = (size_t)tok * D_MODEL;
  hf[base + d0] = o0; hf[base + d1] = o1;
  hb[base + d0] = f2bf(o0); hb[base + d1] = f2bf(o1);
}

// Residual add + LayerNorm:  h = ln(h + add).  grid = NTOK, 256 threads.
__global__ void k_residual_ln(float* __restrict__ hf,
                              const float* __restrict__ add,
                              __bf16* __restrict__ hb) {
  __shared__ float red[256];
  int tok = blockIdx.x;
  size_t base = (size_t)tok * D_MODEL;
  int d0 = threadIdx.x, d1 = threadIdx.x + 256;
  float v0 = hf[base + d0] + add[base + d0];
  float v1 = hf[base + d1] + add[base + d1];
  float mu = block_sum256(v0 + v1, red) * (1.0f / D_MODEL);
  float a0 = v0 - mu, a1 = v1 - mu;
  float var = block_sum256(a0 * a0 + a1 * a1, red) * (1.0f / D_MODEL);
  float rs = rsqrtf(var + 1e-5f);
  float o0 = a0 * rs, o1 = a1 * rs;
  hf[base + d0] = o0; hf[base + d1] = o1;
  hb[base + d0] = f2bf(o0); hb[base + d1] = f2bf(o1);
}

// Causal masked exp + row-normalize (reference "softmax"):
// attn = where(o<=t, exp(min(a,50)), 0);  attn /= (sum + 1e-10)
__global__ void k_attn_norm_row(const float* __restrict__ attn_f,
                                __bf16* __restrict__ attn_b) {
  __shared__ float red[256];
  int t = blockIdx.x;
  const float* row = attn_f + (size_t)t * TT;
  __bf16* orow = attn_b + (size_t)t * TT;
  float vals[4];
  float part = 0.0f;
#pragma unroll
  for (int i = 0; i < 4; ++i) {
    int o = threadIdx.x + i * 256;
    float a = row[o];
    float e = (o <= t) ? expf(fminf(a, 50.0f)) : 0.0f;
    vals[i] = e;
    part += e;
  }
  float s = block_sum256(part, red);
  float inv = 1.0f / (s + 1e-10f);
#pragma unroll
  for (int i = 0; i < 4; ++i) {
    int o = threadIdx.x + i * 256;
    orow[o] = f2bf(vals[i] * inv);
  }
}

__global__ void k_zero_f32(float* __restrict__ p, int n) {
  int i = blockIdx.x * blockDim.x + threadIdx.x;
  if (i < n) p[i] = 0.0f;
}

// ---------------------------------------------------------------------------
// WMMA bf16 GEMM:  C(MxN) = A(MxK) * Bt(NxK)^T   (Bt stored N-major = B^T)
//
// Block = 128 threads = 4 wave32s stacked along M; block tile 256x64.
// Each wave computes a 64x64 macro-tile = 4x4 WMMA tiles (16 v8f accumulators):
//   - per K-step (32): 4 A fragments (direct global b128, private rows) and
//     4 B fragments read from an LDS-staged 64x32 panel shared by all 4 waves.
//   - B panel is double-buffered and filled with GLOBAL_LOAD_ASYNC_TO_LDS_B128
//     (s_wait_asynccnt + barrier), overlapping fetch with 16 WMMAs of compute.
// flags: bit0 = accumulate into Cf, bit1 = relu, bit2 = store bf16 transposed
// ---------------------------------------------------------------------------
#define GF_ACC 1
#define GF_RELU 2
#define GF_STORET 4

__global__ __launch_bounds__(128) void k_gemm_bf16(
    const __bf16* __restrict__ A, int lda,
    const __bf16* __restrict__ Bt, int ldb,
    float* __restrict__ Cf, int ldc,
    __bf16* __restrict__ Cb, int ldcb,
    const float* __restrict__ bias, int K, int flags) {
  __shared__ __bf16 bpan[2][64 * 32];   // double-buffered B panel (4 KB each)

  const int lane = threadIdx.x & 31;
  const int wv = threadIdx.x >> 5;
  const int mblk = blockIdx.y * 256;
  const int nblk = blockIdx.x * 64;
  const int wm0 = mblk + wv * 64;       // this wave's 64-row slab
  const int lr = lane & 15;             // row (A) / col (B,C) within tile
  const int hi = lane >> 4;             // lane group: K sub-range / C row offset
  const int koff = hi * 8;

  v8f acc[4][4];
  {
    v8f z = {};
#pragma unroll
    for (int mt = 0; mt < 4; ++mt)
#pragma unroll
      for (int nt = 0; nt < 4; ++nt) {
        if (flags & GF_ACC) {
#pragma unroll
          for (int i = 0; i < 8; ++i)
            acc[mt][nt][i] =
                Cf[(size_t)(wm0 + mt * 16 + i + hi * 8) * ldc +
                   (nblk + nt * 16 + lr)];
        } else {
          acc[mt][nt] = z;
        }
      }
  }

  // stage one 64x32 B panel into LDS buffer `buf` (2 x 16B chunks per thread)
  auto stage = [&](int buf, int k0) {
#pragma unroll
    for (int c = 0; c < 2; ++c) {
      int chunk = threadIdx.x + c * 128;    // 0..255
      int n = chunk >> 2;                   // panel row 0..63
      int kk = (chunk & 3) * 8;             // 0,8,16,24 (elements)
      __bf16* g = const_cast<__bf16*>(Bt + (size_t)(nblk + n) * ldb + (k0 + kk));
      __bf16* l = &bpan[buf][n * 32 + kk];
#if HAVE_ASYNC_LDS
      __builtin_amdgcn_global_load_async_to_lds_b128(
          (__attribute__((address_space(1))) v4i*)g,
          (__attribute__((address_space(3))) v4i*)l, 0, 0);
#else
      *(v8bf*)l = *(const v8bf*)g;
#endif
    }
  };

  stage(0, 0);
  int cur = 0;
  for (int k0 = 0; k0 < K; k0 += 32) {
    wait_async0();       // this wave's panel chunks have landed in LDS
    __syncthreads();     // every wave's chunks visible; prev reads of ^1 done
    if (k0 + 32 < K) stage(cur ^ 1, k0 + 32);  // overlap next panel fetch

    // A fragments: direct global (per-wave-private rows), ISA 16-bit layout:
    // lanes 0-15 hold K {0..7,16..23}, lanes 16-31 hold K {8..15,24..31}
    v16bf afr[4], bfr[4];
#pragma unroll
    for (int mt = 0; mt < 4; ++mt) {
      const __bf16* ar = A + (size_t)(wm0 + mt * 16 + lr) * lda + k0 + koff;
      if (k0 + 32 < K) __builtin_prefetch(ar + 32, 0, 3);
      v8bf lo = *(const v8bf*)ar;
      v8bf hi8 = *(const v8bf*)(ar + 16);
#pragma unroll
      for (int i = 0; i < 8; ++i) { afr[mt][i] = lo[i]; afr[mt][8 + i] = hi8[i]; }
    }
    // B fragments from the LDS panel (ds_load_b128)
#pragma unroll
    for (int nt = 0; nt < 4; ++nt) {
      const __bf16* br = &bpan[cur][(nt * 16 + lr) * 32 + koff];
      v8bf lo = *(const v8bf*)br;
      v8bf hi8 = *(const v8bf*)(br + 16);
#pragma unroll
      for (int i = 0; i < 8; ++i) { bfr[nt][i] = lo[i]; bfr[nt][8 + i] = hi8[i]; }
    }

    // 16 WMMAs per K-step per wave
#pragma unroll
    for (int mt = 0; mt < 4; ++mt)
#pragma unroll
      for (int nt = 0; nt < 4; ++nt)
        acc[mt][nt] = __builtin_amdgcn_wmma_f32_16x16x32_bf16(
            /*neg_a=*/false, afr[mt], /*neg_b=*/false, bfr[nt],
            /*c_mod=*/(short)0, acc[mt][nt],
            /*reuse_a=*/false, /*reuse_b=*/false);

    __syncthreads();     // all waves done reading bpan[cur] before overwrite
    cur ^= 1;
  }

#pragma unroll
  for (int mt = 0; mt < 4; ++mt) {
#pragma unroll
    for (int nt = 0; nt < 4; ++nt) {
#pragma unroll
      for (int i = 0; i < 8; ++i) {
        int row = wm0 + mt * 16 + i + hi * 8;
        int col = nblk + nt * 16 + lr;
        float v = acc[mt][nt][i];
        if (bias) v += bias[col];
        if (flags & GF_RELU) v = fmaxf(v, 0.0f);
        if (Cf) Cf[(size_t)row * ldc + col] = v;
        if (Cb) {
          __bf16 bv = f2bf(v);
          if (flags & GF_STORET) Cb[(size_t)col * ldcb + row] = bv;
          else                   Cb[(size_t)row * ldcb + col] = bv;
        }
      }
    }
  }
}

// ---------------------------------------------------------------------------
// Host-side orchestration
// ---------------------------------------------------------------------------
extern "C" void kernel_launch(void* const* d_in, const int* in_sizes, int n_in,
                              void* d_out, int out_size, void* d_ws, size_t ws_size,
                              hipStream_t stream) {
  (void)in_sizes; (void)n_in; (void)out_size; (void)ws_size;

  const int*   x    = (const int*)d_in[0];
  // d_in[1] = last_only (False), d_in[2] = capture_stats -> ignored
  const float* emb  = (const float*)d_in[3];
  const float* pos  = (const float*)d_in[4];
  const float* wq   = (const float*)d_in[5];
  const float* wk   = (const float*)d_in[6];
  const float* wvw  = (const float*)d_in[7];
  const float* m0w  = (const float*)d_in[8];
  const float* mbw  = (const float*)d_in[9];
  const float* m1w  = (const float*)d_in[10];
  const float* uew  = (const float*)d_in[11];
  const float* bout = (const float*)d_in[12];
  float* out = (float*)d_out;

  // bump allocator over d_ws
  char* ws = (char*)d_ws;
  size_t off = 0;
  auto alloc = [&](size_t bytes) -> void* {
    off = (off + 255) & ~(size_t)255;
    void* p = ws + off;
    off += bytes;
    return p;
  };

  __bf16* wqT  = (__bf16*)alloc((size_t)N_LAYER * N_HEAD * D_K * D_MODEL * 2);
  __bf16* wkT  = (__bf16*)alloc((size_t)N_LAYER * N_HEAD * D_K * D_MODEL * 2);
  __bf16* wvT  = (__bf16*)alloc((size_t)N_LAYER * N_HEAD * D_MODEL * D_MODEL * 2);
  __bf16* m0T  = (__bf16*)alloc((size_t)N_LAYER * D_HIDDEN * D_MODEL * 2);
  __bf16* m1T  = (__bf16*)alloc((size_t)N_LAYER * D_MODEL * D_HIDDEN * 2);
  __bf16* ueT  = (__bf16*)alloc((size_t)N_DICT * D_MODEL * 2);
  float*  hf   = (float*) alloc((size_t)NTOK * D_MODEL * 4);
  __bf16* hb   = (__bf16*)alloc((size_t)NTOK * D_MODEL * 2);
  __bf16* qb   = (__bf16*)alloc((size_t)N_HEAD * NTOK * D_K * 2);
  __bf16* kb   = (__bf16*)alloc((size_t)N_HEAD * NTOK * D_K * 2);
  __bf16* vT   = (__bf16*)alloc((size_t)N_HEAD * D_MODEL * NTOK * 2);
  float*  attf = (float*) alloc((size_t)TT * TT * 4);
  __bf16* attb = (__bf16*)alloc((size_t)TT * TT * 2);
  float*  vsum = (float*) alloc((size_t)NTOK * D_MODEL * 4);
  __bf16* y1b  = (__bf16*)alloc((size_t)NTOK * D_HIDDEN * 2);
  float*  ymlp = (float*) alloc((size_t)NTOK * D_MODEL * 4);

  auto gemm = [&](const __bf16* A, int lda, const __bf16* Bt, int ldb,
                  float* Cf, int ldc, __bf16* Cb, int ldcb,
                  const float* bias, int M, int N, int K, int flags) {
    dim3 g(N / 64, M / 256, 1);
    k_gemm_bf16<<<g, 128, 0, stream>>>(A, lda, Bt, ldb, Cf, ldc, Cb, ldcb,
                                       bias, K, flags);
  };
  auto xpose = [&](const float* src, __bf16* dst, int R, int C, int batch) {
    dim3 g((C + 15) / 16, (R + 15) / 16, batch);
    k_transpose_to_bf16<<<g, dim3(16, 16, 1), 0, stream>>>(src, dst, R, C);
  };

  // ---- weight conversion (fp32 -> bf16, transposed for N-major B access)
  xpose(wq,  wqT, D_MODEL, D_K,     N_LAYER * N_HEAD);
  xpose(wk,  wkT, D_MODEL, D_K,     N_LAYER * N_HEAD);
  xpose(wvw, wvT, D_MODEL, D_MODEL, N_LAYER * N_HEAD);
  xpose(m0w, m0T, D_MODEL, D_HIDDEN, N_LAYER);
  xpose(m1w, m1T, D_HIDDEN, D_MODEL, N_LAYER);
  xpose(uew, ueT, D_MODEL, N_DICT,  1);

  // ---- embedding + LN
  k_embed_ln<<<NTOK, 256, 0, stream>>>(x, emb, pos, hf, hb);

  // ---- transformer layers
  for (int l = 0; l < N_LAYER; ++l) {
    // projections, per head; V is stored transposed (D_MODEL x NTOK) so the
    // attn*V GEMM gets an N-major B operand with no activation transpose.
    for (int h = 0; h < N_HEAD; ++h) {
      int lh = l * N_HEAD + h;
      gemm(hb, D_MODEL, wqT + (size_t)lh * D_K * D_MODEL, D_MODEL,
           nullptr, 0, qb + (size_t)h * NTOK * D_K, D_K,
           nullptr, NTOK, D_K, D_MODEL, 0);
      gemm(hb, D_MODEL, wkT + (size_t)lh * D_K * D_MODEL, D_MODEL,
           nullptr, 0, kb + (size_t)h * NTOK * D_K, D_K,
           nullptr, NTOK, D_K, D_MODEL, 0);
      gemm(hb, D_MODEL, wvT + (size_t)lh * D_MODEL * D_MODEL, D_MODEL,
           nullptr, 0, vT + (size_t)h * D_MODEL * NTOK, NTOK,
           nullptr, NTOK, D_MODEL, D_MODEL, GF_STORET);
    }

    // vsum accumulates over heads -> zero first
    {
      int n = NTOK * D_MODEL;
      k_zero_f32<<<(n + 255) / 256, 256, 0, stream>>>(vsum, n);
    }

    for (int b = 0; b < BB; ++b) {
      for (int h = 0; h < N_HEAD; ++h) {
        const __bf16* qs = qb + ((size_t)h * NTOK + (size_t)b * TT) * D_K;
        const __bf16* ks = kb + ((size_t)h * NTOK + (size_t)b * TT) * D_K;
        // attn = q @ k^T   (k row-major (o x D_K) is exactly B^T here)
        gemm(qs, D_K, ks, D_K, attf, TT, nullptr, 0, nullptr, TT, TT, D_K, 0);
        k_attn_norm_row<<<TT, 256, 0, stream>>>(attf, attb);
        // vsum[b] += attn @ v_head   (vT slice: ldb = NTOK, col offset b*TT)
        gemm(attb, TT, vT + (size_t)h * D_MODEL * NTOK + (size_t)b * TT, NTOK,
             vsum + (size_t)b * TT * D_MODEL, D_MODEL, nullptr, 0, nullptr,
             TT, D_MODEL, TT, GF_ACC);
      }
    }

    // h = ln(h + vsum)
    k_residual_ln<<<NTOK, 256, 0, stream>>>(hf, vsum, hb);

    // MLP: y1 = relu(h @ m0 + mb);  y = y1 @ m1;  h = ln(h + y)
    gemm(hb, D_MODEL, m0T + (size_t)l * D_HIDDEN * D_MODEL, D_MODEL,
         nullptr, 0, y1b, D_HIDDEN, mbw + (size_t)l * D_HIDDEN,
         NTOK, D_HIDDEN, D_MODEL, GF_RELU);
    gemm(y1b, D_HIDDEN, m1T + (size_t)l * D_MODEL * D_HIDDEN, D_HIDDEN,
         ymlp, D_MODEL, nullptr, 0, nullptr, NTOK, D_MODEL, D_HIDDEN, 0);
    k_residual_ln<<<NTOK, 256, 0, stream>>>(hf, ymlp, hb);
  }

  // ---- unembedding + bias -> logits (fp32 out)
  gemm(hb, D_MODEL, ueT, D_MODEL, out, N_DICT, nullptr, 0, bout,
       NTOK, N_DICT, D_MODEL, 0);
}
